// GnBlock_36532991820470
// MI455X (gfx1250) — compile-verified, matrix operand order checked by reading, not run on hardware
//
#include <hip/hip_runtime.h>

// ---------------------------------------------------------------------------
// GraphNet block on MI455X (gfx1250, wave32, WMMA bf16 16x16x32)
// Each wave owns TWO 16-row M-tiles so every B fragment feeds 2 WMMAs.
// ---------------------------------------------------------------------------

#define H 128
#define N_NODES 100000
#define N_EDGES 500000
#define LDSP 136          // padded LDS row stride (ushorts): 272B, 16B-aligned

typedef __attribute__((ext_vector_type(16))) __bf16 v16bf;
typedef __attribute__((ext_vector_type(8)))  float  v8f;

union BFrag {
    unsigned u[8];
    v16bf    v;
};

// cheap fp32 -> bf16 (round-half-up) for scalar stores
__device__ __forceinline__ unsigned short f2bf(float f) {
    return (unsigned short)((__float_as_uint(f) + 0x8000u) >> 16);
}
// pack two fp32 -> packed bf16 pair with one v_perm_b32
__device__ __forceinline__ unsigned pack2(float lo, float hi) {
    unsigned ul = __float_as_uint(lo) + 0x8000u;
    unsigned uh = __float_as_uint(hi) + 0x8000u;
    return __builtin_amdgcn_perm(uh, ul, 0x07060302u);  // {uh[31:16], ul[31:16]}
}

// A-fragment (16x32 bf16) gathered from a global fp32 row (this lane's row m)
// pairs j=0..3 <- floats [colbase+half*8 .. +7]; j=4..7 <- +16
__device__ __forceinline__ void load_a_global(BFrag& f, const float* row,
                                              int colbase, int half) {
    const float* p = row + colbase + half * 8;
    float4 p0 = *(const float4*)(p);
    float4 p1 = *(const float4*)(p + 4);
    float4 p2 = *(const float4*)(p + 16);
    float4 p3 = *(const float4*)(p + 20);
    f.u[0] = pack2(p0.x, p0.y); f.u[1] = pack2(p0.z, p0.w);
    f.u[2] = pack2(p1.x, p1.y); f.u[3] = pack2(p1.z, p1.w);
    f.u[4] = pack2(p2.x, p2.y); f.u[5] = pack2(p2.z, p2.w);
    f.u[6] = pack2(p3.x, p3.y); f.u[7] = pack2(p3.z, p3.w);
}

// B-fragment (32x16 bf16) from transposed bf16 weights: wt_row = Wt + n*K
__device__ __forceinline__ void load_b(BFrag& f, const unsigned short* wt_row,
                                       int kbase, int half) {
    uint4 q0 = *(const uint4*)(wt_row + kbase + half * 8);
    uint4 q1 = *(const uint4*)(wt_row + kbase + 16 + half * 8);
    f.u[0] = q0.x; f.u[1] = q0.y; f.u[2] = q0.z; f.u[3] = q0.w;
    f.u[4] = q1.x; f.u[5] = q1.y; f.u[6] = q1.z; f.u[7] = q1.w;
}

// A-fragment from the per-wave LDS activation tile (row m, padded stride LDSP)
__device__ __forceinline__ void load_a_lds(BFrag& f, const unsigned short* hrow,
                                           int kbase, int half) {
    uint4 q0 = *(const uint4*)(hrow + kbase + half * 8);
    uint4 q1 = *(const uint4*)(hrow + kbase + 16 + half * 8);
    f.u[0] = q0.x; f.u[1] = q0.y; f.u[2] = q0.z; f.u[3] = q0.w;
    f.u[4] = q1.x; f.u[5] = q1.y; f.u[6] = q1.z; f.u[7] = q1.w;
}

__device__ __forceinline__ v8f wmma_bf16(const BFrag& a, const BFrag& b, v8f c) {
    return __builtin_amdgcn_wmma_f32_16x16x32_bf16(
        /*neg_a=*/false, a.v, /*neg_b=*/false, b.v,
        /*c_mod=*/(short)0, c, /*reuse_a=*/false, /*reuse_b=*/false);
}

// bias (+optional ReLU) then stash 16x128 tile as bf16 in LDS (C-layout -> row-major)
__device__ __forceinline__ void bias_act_store(const v8f* acc, const float* bias,
                                               unsigned short (*hb)[LDSP],
                                               int nl, int half, bool relu) {
#pragma unroll
    for (int nt = 0; nt < 8; ++nt) {
        float bn = bias[nt * 16 + nl];
#pragma unroll
        for (int r = 0; r < 8; ++r) {
            float v = acc[nt][r] + bn;
            if (relu) v = fmaxf(v, 0.0f);
            hb[r + 8 * half][nt * 16 + nl] = f2bf(v);
        }
    }
}

// one 128->128 layer for both M-tiles, activations from LDS
__device__ __forceinline__ void layer128_dual(v8f (*acc)[8], const unsigned short* wt,
                                              const unsigned short* h0,
                                              const unsigned short* h1,
                                              int nl, int half) {
    v8f z = {};
#pragma unroll
    for (int nt = 0; nt < 8; ++nt) { acc[0][nt] = z; acc[1][nt] = z; }
    for (int kt = 0; kt < 4; ++kt) {
        BFrag a0, a1;
        load_a_lds(a0, h0, kt * 32, half);
        load_a_lds(a1, h1, kt * 32, half);
#pragma unroll
        for (int nt = 0; nt < 8; ++nt) {
            BFrag b;
            load_b(b, wt + (size_t)(nt * 16 + nl) * 128, kt * 32, half);
            acc[0][nt] = wmma_bf16(a0, b, acc[0][nt]);
            acc[1][nt] = wmma_bf16(a1, b, acc[1][nt]);
        }
    }
}

// ---------------------------------------------------------------------------
// Edge block: e_new = LN(MLP([x_s | x_r | ea])); out_e = ea + e_new; agg += e_new
// ---------------------------------------------------------------------------
__global__ void __launch_bounds__(128) edge_kernel(
    const float* __restrict__ x, const float* __restrict__ ea,
    const long long* __restrict__ eidx,
    const unsigned short* __restrict__ W1t, const unsigned short* __restrict__ W2t,
    const unsigned short* __restrict__ W3t,
    const float* __restrict__ b1, const float* __restrict__ b2,
    const float* __restrict__ b3, const float* __restrict__ g,
    const float* __restrict__ beta,
    float* __restrict__ out_e, float* __restrict__ agg)
{
    __shared__ __align__(16) unsigned short hbuf[4][2][16][LDSP];

    const int wave = threadIdx.x >> 5;
    const int lane = threadIdx.x & 31;
    const int half = lane >> 4;
    const int nl   = lane & 15;
    const int tile = blockIdx.x * 4 + wave;
    if (tile >= N_EDGES / 32) return;
    const int e0 = tile * 32;
    const int m  = nl;  // A-matrix row owned by this lane (per tile)

    // gather pointers: rowsA[t] = this lane's row of e_in for M-tile t
    const float* rowsA[2][3];
#pragma unroll
    for (int t = 0; t < 2; ++t) {
        int e = e0 + t * 16 + m;
        long long sIdx = eidx[e];
        long long rIdx = eidx[N_EDGES + e];
        rowsA[t][0] = x + (size_t)sIdx * H;
        rowsA[t][1] = x + (size_t)rIdx * H;
        rowsA[t][2] = ea + (size_t)e * H;
    }
    if (e0 + 128 + m < N_EDGES)
        __builtin_prefetch(ea + (size_t)(e0 + 128 + m) * H, 0, 1);

    v8f acc[2][8];
    v8f z = {};
#pragma unroll
    for (int nt = 0; nt < 8; ++nt) { acc[0][nt] = z; acc[1][nt] = z; }

    // ---- layer 1: 384 -> 128 ----
    for (int kt = 0; kt < 12; ++kt) {
        BFrag a0, a1;
        load_a_global(a0, rowsA[0][kt >> 2], (kt & 3) * 32, half);
        load_a_global(a1, rowsA[1][kt >> 2], (kt & 3) * 32, half);
#pragma unroll
        for (int nt = 0; nt < 8; ++nt) {
            BFrag b;
            load_b(b, W1t + (size_t)(nt * 16 + nl) * 384, kt * 32, half);
            acc[0][nt] = wmma_bf16(a0, b, acc[0][nt]);
            acc[1][nt] = wmma_bf16(a1, b, acc[1][nt]);
        }
    }
    bias_act_store(acc[0], b1, hbuf[wave][0], nl, half, true);
    bias_act_store(acc[1], b1, hbuf[wave][1], nl, half, true);
    asm volatile("s_wait_dscnt 0" ::: "memory");

    // ---- layer 2 ----
    layer128_dual(acc, W2t, &hbuf[wave][0][m][0], &hbuf[wave][1][m][0], nl, half);
    bias_act_store(acc[0], b2, hbuf[wave][0], nl, half, true);
    bias_act_store(acc[1], b2, hbuf[wave][1], nl, half, true);
    asm volatile("s_wait_dscnt 0" ::: "memory");

    // ---- layer 3 (no ReLU) ----
    layer128_dual(acc, W3t, &hbuf[wave][0][m][0], &hbuf[wave][1][m][0], nl, half);

    // ---- LayerNorm + residual + scatter, per M-tile ----
    float b3v[8], gv[8], bev[8];
#pragma unroll
    for (int nt = 0; nt < 8; ++nt) {
        int n = nt * 16 + nl;
        b3v[nt] = b3[n]; gv[nt] = g[n]; bev[nt] = beta[n];
    }
#pragma unroll
    for (int t = 0; t < 2; ++t) {
        float s8[8] = {0, 0, 0, 0, 0, 0, 0, 0};
        float q8[8] = {0, 0, 0, 0, 0, 0, 0, 0};
#pragma unroll
        for (int nt = 0; nt < 8; ++nt)
#pragma unroll
            for (int r = 0; r < 8; ++r) {
                float v = acc[t][nt][r] + b3v[nt];
                s8[r] += v; q8[r] += v * v;
            }
#pragma unroll
        for (int off = 1; off < 16; off <<= 1)
#pragma unroll
            for (int r = 0; r < 8; ++r) {
                s8[r] += __shfl_xor(s8[r], off, 32);
                q8[r] += __shfl_xor(q8[r], off, 32);
            }
#pragma unroll
        for (int r = 0; r < 8; ++r) {
            float muv = s8[r] * (1.0f / H);
            float var = q8[r] * (1.0f / H) - muv * muv;
            float rnv = rsqrtf(var + 1e-5f);
            int e = e0 + t * 16 + r + 8 * half;
            long long rcv = eidx[N_EDGES + e];
            const float* ea_row = ea + (size_t)e * H;
            float* oe_row = out_e + (size_t)e * H;
            float* ag_row = agg + (size_t)rcv * H;
#pragma unroll
            for (int nt = 0; nt < 8; ++nt) {
                int n = nt * 16 + nl;
                float v = acc[t][nt][r] + b3v[nt];
                float d = (v - muv) * rnv * gv[nt] + bev[nt];
                oe_row[n] = ea_row[n] + d;
                atomicAdd(ag_row + n, d);
            }
        }
    }
}

// ---------------------------------------------------------------------------
// Node block: x_new = LN(MLP([x | agg])); out_x = x + x_new
// ---------------------------------------------------------------------------
__global__ void __launch_bounds__(128) node_kernel(
    const float* __restrict__ x, const float* __restrict__ agg,
    const unsigned short* __restrict__ W1t, const unsigned short* __restrict__ W2t,
    const unsigned short* __restrict__ W3t,
    const float* __restrict__ b1, const float* __restrict__ b2,
    const float* __restrict__ b3, const float* __restrict__ g,
    const float* __restrict__ beta,
    float* __restrict__ out_x)
{
    __shared__ __align__(16) unsigned short hbuf[4][2][16][LDSP];

    const int wave = threadIdx.x >> 5;
    const int lane = threadIdx.x & 31;
    const int half = lane >> 4;
    const int nl   = lane & 15;
    const int tile = blockIdx.x * 4 + wave;
    if (tile >= N_NODES / 32) return;
    const int v0 = tile * 32;
    const int m  = nl;

    const float* rowsA[2][2];
#pragma unroll
    for (int t = 0; t < 2; ++t) {
        int v = v0 + t * 16 + m;
        rowsA[t][0] = x + (size_t)v * H;
        rowsA[t][1] = agg + (size_t)v * H;
    }

    v8f acc[2][8];
    v8f z = {};
#pragma unroll
    for (int nt = 0; nt < 8; ++nt) { acc[0][nt] = z; acc[1][nt] = z; }

    // ---- layer 1: 256 -> 128 ----
    for (int kt = 0; kt < 8; ++kt) {
        BFrag a0, a1;
        load_a_global(a0, rowsA[0][kt >> 2], (kt & 3) * 32, half);
        load_a_global(a1, rowsA[1][kt >> 2], (kt & 3) * 32, half);
#pragma unroll
        for (int nt = 0; nt < 8; ++nt) {
            BFrag b;
            load_b(b, W1t + (size_t)(nt * 16 + nl) * 256, kt * 32, half);
            acc[0][nt] = wmma_bf16(a0, b, acc[0][nt]);
            acc[1][nt] = wmma_bf16(a1, b, acc[1][nt]);
        }
    }
    bias_act_store(acc[0], b1, hbuf[wave][0], nl, half, true);
    bias_act_store(acc[1], b1, hbuf[wave][1], nl, half, true);
    asm volatile("s_wait_dscnt 0" ::: "memory");

    layer128_dual(acc, W2t, &hbuf[wave][0][m][0], &hbuf[wave][1][m][0], nl, half);
    bias_act_store(acc[0], b2, hbuf[wave][0], nl, half, true);
    bias_act_store(acc[1], b2, hbuf[wave][1], nl, half, true);
    asm volatile("s_wait_dscnt 0" ::: "memory");

    layer128_dual(acc, W3t, &hbuf[wave][0][m][0], &hbuf[wave][1][m][0], nl, half);

    float b3v[8], gv[8], bev[8];
#pragma unroll
    for (int nt = 0; nt < 8; ++nt) {
        int n = nt * 16 + nl;
        b3v[nt] = b3[n]; gv[nt] = g[n]; bev[nt] = beta[n];
    }
#pragma unroll
    for (int t = 0; t < 2; ++t) {
        float s8[8] = {0, 0, 0, 0, 0, 0, 0, 0};
        float q8[8] = {0, 0, 0, 0, 0, 0, 0, 0};
#pragma unroll
        for (int nt = 0; nt < 8; ++nt)
#pragma unroll
            for (int r = 0; r < 8; ++r) {
                float v = acc[t][nt][r] + b3v[nt];
                s8[r] += v; q8[r] += v * v;
            }
#pragma unroll
        for (int off = 1; off < 16; off <<= 1)
#pragma unroll
            for (int r = 0; r < 8; ++r) {
                s8[r] += __shfl_xor(s8[r], off, 32);
                q8[r] += __shfl_xor(q8[r], off, 32);
            }
#pragma unroll
        for (int r = 0; r < 8; ++r) {
            float muv = s8[r] * (1.0f / H);
            float var = q8[r] * (1.0f / H) - muv * muv;
            float rnv = rsqrtf(var + 1e-5f);
            int vtx = v0 + t * 16 + r + 8 * half;
            const float* x_row = x + (size_t)vtx * H;
            float* ox_row = out_x + (size_t)vtx * H;
#pragma unroll
            for (int nt = 0; nt < 8; ++nt) {
                int n = nt * 16 + nl;
                float v = acc[t][nt][r] + b3v[nt];
                float d = (v - muv) * rnv * gv[nt] + bev[nt];
                ox_row[n] = x_row[n] + d;
            }
        }
    }
}

// ---------------------------------------------------------------------------
// helpers: fp32 -> transposed bf16 weights; zero the aggregation buffer
// ---------------------------------------------------------------------------
__global__ void prep_w(const float* __restrict__ w, unsigned short* __restrict__ wt, int K) {
    int i = blockIdx.x * blockDim.x + threadIdx.x;
    if (i < K * H) {
        int k = i / H, n = i % H;
        wt[(size_t)n * K + k] = f2bf(w[i]);
    }
}

__global__ void zero_f32(float* __restrict__ p, int n) {
    int i = blockIdx.x * blockDim.x + threadIdx.x;
    if (i < n) p[i] = 0.0f;
}

// ---------------------------------------------------------------------------
extern "C" void kernel_launch(void* const* d_in, const int* in_sizes, int n_in,
                              void* d_out, int out_size, void* d_ws, size_t ws_size,
                              hipStream_t stream) {
    const float*     x    = (const float*)d_in[0];
    const float*     ea   = (const float*)d_in[1];
    const long long* eidx = (const long long*)d_in[2];
    const float *eW1 = (const float*)d_in[3],  *eb1 = (const float*)d_in[4];
    const float *eW2 = (const float*)d_in[5],  *eb2 = (const float*)d_in[6];
    const float *eW3 = (const float*)d_in[7],  *eb3 = (const float*)d_in[8];
    const float *eg  = (const float*)d_in[9],  *ebt = (const float*)d_in[10];
    const float *nW1 = (const float*)d_in[11], *nb1 = (const float*)d_in[12];
    const float *nW2 = (const float*)d_in[13], *nb2 = (const float*)d_in[14];
    const float *nW3 = (const float*)d_in[15], *nb3 = (const float*)d_in[16];
    const float *ng  = (const float*)d_in[17], *nbt = (const float*)d_in[18];

    float* out_x = (float*)d_out;                       // N_NODES * H
    float* out_e = out_x + (size_t)N_NODES * H;         // N_EDGES * H

    // workspace: [agg fp32 N_NODES*H][bf16 transposed weights]
    float* agg = (float*)d_ws;
    unsigned short* wb = (unsigned short*)((char*)d_ws + (size_t)N_NODES * H * sizeof(float));
    unsigned short* eW1t = wb;
    unsigned short* eW2t = eW1t + (size_t)H * 384;
    unsigned short* eW3t = eW2t + (size_t)H * 128;
    unsigned short* nW1t = eW3t + (size_t)H * 128;
    unsigned short* nW2t = nW1t + (size_t)H * 256;
    unsigned short* nW3t = nW2t + (size_t)H * 128;

    // 1) convert/transpose weights to bf16 (tiny)
    prep_w<<<(384 * H + 255) / 256, 256, 0, stream>>>(eW1, eW1t, 384);
    prep_w<<<(128 * H + 255) / 256, 256, 0, stream>>>(eW2, eW2t, 128);
    prep_w<<<(128 * H + 255) / 256, 256, 0, stream>>>(eW3, eW3t, 128);
    prep_w<<<(256 * H + 255) / 256, 256, 0, stream>>>(nW1, nW1t, 256);
    prep_w<<<(128 * H + 255) / 256, 256, 0, stream>>>(nW2, nW2t, 128);
    prep_w<<<(128 * H + 255) / 256, 256, 0, stream>>>(nW3, nW3t, 128);

    // 2) zero aggregation buffer (fits in L2, atomics land there)
    int aggN = N_NODES * H;
    zero_f32<<<(aggN + 255) / 256, 256, 0, stream>>>(agg, aggN);

    // 3) edge block (writes out_e, scatters into agg)
    int etiles = N_EDGES / 32;
    edge_kernel<<<(etiles + 3) / 4, 128, 0, stream>>>(
        x, ea, eidx, eW1t, eW2t, eW3t, eb1, eb2, eb3, eg, ebt, out_e, agg);

    // 4) node block (reads agg, writes out_x)
    int ntiles = N_NODES / 32;
    node_kernel<<<(ntiles + 3) / 4, 128, 0, stream>>>(
        x, agg, nW1t, nW2t, nW3t, nb1, nb2, nb3, ng, nbt, out_x);
}
